// LoRALinear_8744553415010
// MI455X (gfx1250) — matrix operand order repaired and checked by reading, not compiled, not run on hardware
//
#include <hip/hip_runtime.h>

// LoRA linear: out = x*W^T + bias + 2*(x*A^T)*B^T
// bf16x3 split-GEMM on v_wmma_f32_16x16x32_bf16, fp32 accumulate.
// Fast path: one-shot fp32->bf16(hi/lo) convert pass (W stored pre-transposed
// K-major), then a TDM-fed double-buffered GEMM: tensor_load_to_lds stages
// padded tiles into LDS, synchronized on TENSORcnt.
// Fallback (small workspace): in-kernel convert GEMM (round-1 path).

#define TOKENS 8192
#define DIN    4096
#define DOUT   4096
#define RANK   16

#define BM 128
#define BN 128
#define BK 32
#define LDXK 40    // x-tile LDS row stride in bf16 (32 + 8 pad -> 80B rows)
#define LDWN 136   // w-tile (K-major) LDS row stride in bf16 (128 + 8 pad -> 272B rows)

typedef __attribute__((ext_vector_type(16))) __bf16 v16bf;
typedef __attribute__((ext_vector_type(8)))  float  v8f;
typedef __attribute__((ext_vector_type(4)))  unsigned int ui32x4;
typedef __attribute__((ext_vector_type(4)))  int i32x4;
typedef __attribute__((ext_vector_type(8)))  int i32x8;

__device__ __forceinline__ unsigned short f2bf(float f) {
    unsigned int u = __float_as_uint(f);
    u += 0x7FFFu + ((u >> 16) & 1u);          // round-to-nearest-even
    return (unsigned short)(u >> 16);
}
__device__ __forceinline__ float bf2f(unsigned short h) {
    return __uint_as_float(((unsigned int)h) << 16);
}

__device__ __forceinline__ v16bf make_frag(const unsigned short* p0, const unsigned short* p1) {
    v16bf f;
    ((uint4*)&f)[0] = *(const uint4*)p0;
    ((uint4*)&f)[1] = *(const uint4*)p1;
    return f;
}

// ---------------------------------------------------------------------------
// TDM helpers (CDNA5 ISA): 2-D tile, 2-byte elements,
// LDS row padding via pad_enable/pad_interval/pad_amount.
// ---------------------------------------------------------------------------
__device__ __forceinline__ void tdm_load_2d(unsigned lds_off,
                                            const unsigned short* gptr,
                                            unsigned tile_d0, unsigned tile_d1,
                                            unsigned stride_elems,
                                            unsigned pad_interval, unsigned pad_amount)
{
#if defined(__gfx1250__) && __has_builtin(__builtin_amdgcn_tensor_load_to_lds)
    unsigned long long ga = (unsigned long long)(const void*)gptr;
    ui32x4 g0;
    g0[0] = 1u;                                   // count=1, user descriptor
    g0[1] = lds_off;                              // LDS byte address (wave-relative)
    g0[2] = (unsigned)ga;                         // global_addr[31:0]
    g0[3] = ((unsigned)(ga >> 32) & 0x01FFFFFFu)  // global_addr[56:32]
            | 0x80000000u;                        // type = 2 ("image")
    i32x8 g1;
    g1[0] = (int)((1u << 16)                      // data_size = 2 bytes
                  | (1u << 20)                    // pad_enable
                  | (pad_interval << 22)
                  | (pad_amount << 25));
    g1[1] = (int)((tile_d0 & 0xFFFFu) << 16);                                  // tensor_dim0 lo
    g1[2] = (int)(((tile_d0 >> 16) & 0xFFFFu) | ((tile_d1 & 0xFFFFu) << 16));  // td0 hi | td1 lo
    g1[3] = (int)(((tile_d1 >> 16) & 0xFFFFu) | ((tile_d0 & 0xFFFFu) << 16));  // td1 hi | tile_dim0
    g1[4] = (int)(tile_d1 & 0xFFFFu);                                          // tile_dim1 | tile_dim2=0
    g1[5] = (int)stride_elems;                                                 // tensor_dim0_stride[31:0]
    g1[6] = 0;                                                                 // stride hi | dim1_stride lo
    g1[7] = 0;
    i32x4 z4; z4[0] = 0; z4[1] = 0; z4[2] = 0; z4[3] = 0;
#if __clang_major__ >= 23
    i32x8 z8; z8[0]=0; z8[1]=0; z8[2]=0; z8[3]=0; z8[4]=0; z8[5]=0; z8[6]=0; z8[7]=0;
    __builtin_amdgcn_tensor_load_to_lds(g0, g1, z4, z4, z8, 0);
#else
    __builtin_amdgcn_tensor_load_to_lds(g0, g1, z4, z4, 0);
#endif
#else
    (void)lds_off; (void)gptr; (void)tile_d0; (void)tile_d1;
    (void)stride_elems; (void)pad_interval; (void)pad_amount;
#endif
}

__device__ __forceinline__ void wait_tensor0() {
#if defined(__gfx1250__) && __has_builtin(__builtin_amdgcn_s_wait_tensorcnt)
    __builtin_amdgcn_s_wait_tensorcnt(0);
#endif
}

// ---------------------------------------------------------------------------
// Pre-pass 1: x -> x_hi/x_lo bf16, same [T][DIN] layout.
// ---------------------------------------------------------------------------
__global__ __launch_bounds__(256)
void convert_x_kernel(const float* __restrict__ in,
                      unsigned short* __restrict__ oh,
                      unsigned short* __restrict__ ol)
{
    const size_t idx = ((size_t)blockIdx.x * 256 + threadIdx.x) * 4;
    const float4 v = *(const float4*)(in + idx);
    float f[4] = {v.x, v.y, v.z, v.w};
    unsigned short h[4], l[4];
    #pragma unroll
    for (int j = 0; j < 4; ++j) {
        h[j] = f2bf(f[j]);
        l[j] = f2bf(f[j] - bf2f(h[j]));
    }
    uint2 ph, pl;
    ph.x = (unsigned)h[0] | ((unsigned)h[1] << 16);
    ph.y = (unsigned)h[2] | ((unsigned)h[3] << 16);
    pl.x = (unsigned)l[0] | ((unsigned)l[1] << 16);
    pl.y = (unsigned)l[2] | ((unsigned)l[3] << 16);
    *(uint2*)(oh + idx) = ph;
    *(uint2*)(ol + idx) = pl;
}

// ---------------------------------------------------------------------------
// Pre-pass 2: W[DOUT][DIN] -> wT_hi/wT_lo bf16 [DIN][DOUT] (K-major for the
// WMMA B operand: lane = N, elements = K). LDS-tiled transpose, coalesced.
// ---------------------------------------------------------------------------
#define TT 64
__global__ __launch_bounds__(256)
void convert_wT_kernel(const float* __restrict__ w,
                       unsigned short* __restrict__ oh,
                       unsigned short* __restrict__ ol)
{
    __shared__ unsigned short shi[TT][TT + 8];
    __shared__ unsigned short slo[TT][TT + 8];
    const int n0  = blockIdx.x * TT;          // W row block  (wT column block)
    const int k0  = blockIdx.y * TT;          // W col block  (wT row block)
    const int tr  = threadIdx.x >> 4;         // 0..15
    const int tc4 = (threadIdx.x & 15) << 2;  // 0..60 step 4

    #pragma unroll
    for (int i = 0; i < 4; ++i) {
        const int n = tr + 16 * i;
        const float4 v = *(const float4*)(w + (size_t)(n0 + n) * DIN + k0 + tc4);
        float f[4] = {v.x, v.y, v.z, v.w};
        #pragma unroll
        for (int j = 0; j < 4; ++j) {
            const unsigned short h = f2bf(f[j]);
            shi[tc4 + j][n] = h;
            slo[tc4 + j][n] = f2bf(f[j] - bf2f(h));
        }
    }
    __syncthreads();
    #pragma unroll
    for (int i = 0; i < 4; ++i) {
        const int k = tr + 16 * i;
        const unsigned short* ph = &shi[k][tc4];
        const unsigned short* pl = &slo[k][tc4];
        uint2 vh, vl;
        vh.x = (unsigned)ph[0] | ((unsigned)ph[1] << 16);
        vh.y = (unsigned)ph[2] | ((unsigned)ph[3] << 16);
        vl.x = (unsigned)pl[0] | ((unsigned)pl[1] << 16);
        vl.y = (unsigned)pl[2] | ((unsigned)pl[3] << 16);
        *(uint2*)(oh + (size_t)(k0 + k) * DOUT + n0 + tc4) = vh;
        *(uint2*)(ol + (size_t)(k0 + k) * DOUT + n0 + tc4) = vl;
    }
}

// ---------------------------------------------------------------------------
// txA[t][r] = 2 * sum_k x[t][k] * A[r][k]   (SCALE folded in; exact fp32)
// ---------------------------------------------------------------------------
__global__ __launch_bounds__(256)
void lora_xa_kernel(const float* __restrict__ x, const float* __restrict__ loraA,
                    float* __restrict__ txA)
{
    const int lane  = threadIdx.x & 31;
    const int wave  = threadIdx.x >> 5;
    const int t     = blockIdx.x * 8 + wave;
    const int r     = lane & 15;
    const int khalf = lane >> 4;
    const float* xrow = x + (size_t)t * DIN;
    const float* arow = loraA + (size_t)r * DIN;
    float acc = 0.0f;
    for (int k = khalf * 4; k < DIN; k += 8) {
        float4 xv = *(const float4*)(xrow + k);
        float4 av = *(const float4*)(arow + k);
        acc += xv.x * av.x + xv.y * av.y + xv.z * av.z + xv.w * av.w;
    }
    acc += __shfl_xor(acc, 16, 32);
    if (lane < 16) txA[(size_t)t * RANK + lane] = 2.0f * acc;
}

// ---------------------------------------------------------------------------
// Shared GEMM pieces
// ---------------------------------------------------------------------------
struct WaveCtx {
    int tid, lane, wave, wm, wn, bm0, bn0;
};

__device__ __forceinline__ void gemm_epilogue(const WaveCtx& c, v8f acc[4][2],
                                              const float* __restrict__ bias,
                                              const float* __restrict__ loraB,
                                              const float* __restrict__ txA,
                                              float* __restrict__ out)
{
    const int lane15 = c.lane & 15;
    const int mhalf  = (c.lane >> 4) << 3;
    #pragma unroll
    for (int ni = 0; ni < 2; ++ni) {
        const int o = c.bn0 + c.wn * 32 + ni * 16 + lane15;
        const float bsum = bias[o];
        const float4* Bo = (const float4*)(loraB + (size_t)o * RANK);
        const float4 b0 = Bo[0], b1 = Bo[1], b2 = Bo[2], b3 = Bo[3];
        #pragma unroll
        for (int mi = 0; mi < 4; ++mi) {
            const int t0 = c.bm0 + c.wm * 64 + mi * 16 + mhalf;
            #pragma unroll
            for (int j = 0; j < 8; ++j) {
                const int t = t0 + j;
                const float4* Ta = (const float4*)(txA + (size_t)t * RANK);
                const float4 a0 = Ta[0], a1 = Ta[1], a2 = Ta[2], a3 = Ta[3];
                const float dotv =
                    a0.x * b0.x + a0.y * b0.y + a0.z * b0.z + a0.w * b0.w +
                    a1.x * b1.x + a1.y * b1.y + a1.z * b1.z + a1.w * b1.w +
                    a2.x * b2.x + a2.y * b2.y + a2.z * b2.z + a2.w * b2.w +
                    a3.x * b3.x + a3.y * b3.y + a3.z * b3.z + a3.w * b3.w;
                out[(size_t)t * DOUT + o] = acc[mi][ni][j] + bsum + dotv;
            }
        }
    }
}

__device__ __forceinline__ void gemm_mma_step(const WaveCtx& c,
                                              const unsigned short* sxh, const unsigned short* sxl,
                                              const unsigned short* swh, const unsigned short* swl,
                                              v8f acc[4][2])
{
    v16bf ah[4], al[4], bh[2], bl[2];
    const int ko = (c.lane >> 4) << 3;
    #pragma unroll
    for (int mi = 0; mi < 4; ++mi) {
        const int row = c.wm * 64 + mi * 16 + (c.lane & 15);
        const unsigned short* ph = sxh + row * LDXK + ko;
        const unsigned short* pl = sxl + row * LDXK + ko;
        ah[mi] = make_frag(ph, ph + 16);
        al[mi] = make_frag(pl, pl + 16);
    }
    #pragma unroll
    for (int ni = 0; ni < 2; ++ni) {
        const int nb = c.wn * 32 + ni * 16;
        const unsigned short* ph = swh + nb + (c.lane & 15);
        const unsigned short* pl = swl + nb + (c.lane & 15);
        #pragma unroll
        for (int s = 0; s < 16; ++s) {
            const int k = ko + s + (s >= 8 ? 8 : 0);
            bh[ni][s] = __builtin_bit_cast(__bf16, ph[k * LDWN]);
            bl[ni][s] = __builtin_bit_cast(__bf16, pl[k * LDWN]);
        }
    }
    #pragma unroll
    for (int mi = 0; mi < 4; ++mi)
        #pragma unroll
        for (int ni = 0; ni < 2; ++ni) {
            v8f cc = acc[mi][ni];
            cc = __builtin_amdgcn_wmma_f32_16x16x32_bf16(false, ah[mi], false, bh[ni], (short)0, cc, false, false);
            cc = __builtin_amdgcn_wmma_f32_16x16x32_bf16(false, ah[mi], false, bl[ni], (short)0, cc, false, false);
            cc = __builtin_amdgcn_wmma_f32_16x16x32_bf16(false, al[mi], false, bh[ni], (short)0, cc, false, false);
            acc[mi][ni] = cc;
        }
}

// ---------------------------------------------------------------------------
// Fast path GEMM: TDM-fed, double-buffered on TENSORcnt.
// ---------------------------------------------------------------------------
__global__ __launch_bounds__(256)
void lora_gemm_tdm_kernel(const unsigned short* __restrict__ xh,
                          const unsigned short* __restrict__ xl,
                          const unsigned short* __restrict__ wth,
                          const unsigned short* __restrict__ wtl,
                          const float* __restrict__ bias,
                          const float* __restrict__ loraB,
                          const float* __restrict__ txA,
                          float* __restrict__ out)
{
    __shared__ __align__(16) unsigned short sxh[2][BM * LDXK];
    __shared__ __align__(16) unsigned short sxl[2][BM * LDXK];
    __shared__ __align__(16) unsigned short swh[2][BK * LDWN];
    __shared__ __align__(16) unsigned short swl[2][BK * LDWN];

    WaveCtx c;
    c.tid = threadIdx.x; c.lane = c.tid & 31; c.wave = c.tid >> 5;
    c.wm = c.wave >> 2; c.wn = c.wave & 3;
    c.bm0 = blockIdx.y * BM; c.bn0 = blockIdx.x * BN;

    v8f acc[4][2];
    #pragma unroll
    for (int mi = 0; mi < 4; ++mi)
        #pragma unroll
        for (int ni = 0; ni < 2; ++ni)
            #pragma unroll
            for (int e = 0; e < 8; ++e)
                acc[mi][ni][e] = 0.0f;

    auto issue = [&](int kt, int s) {
        const size_t xoff = (size_t)c.bm0 * DIN + (size_t)kt * BK;
        const size_t woff = (size_t)kt * BK * DOUT + c.bn0;
        // x tile: 32 elems/row (64B) x 128 rows, pad 16B -> 80B LDS rows (LDXK)
        tdm_load_2d((unsigned)(size_t)&sxh[s][0], xh + xoff, BK, BM, DIN, 3u, 3u);
        tdm_load_2d((unsigned)(size_t)&sxl[s][0], xl + xoff, BK, BM, DIN, 3u, 3u);
        // w tile: 128 elems/row (256B) x 32 rows, pad 16B -> 272B LDS rows (LDWN)
        tdm_load_2d((unsigned)(size_t)&swh[s][0], wth + woff, BN, BK, DOUT, 5u, 3u);
        tdm_load_2d((unsigned)(size_t)&swl[s][0], wtl + woff, BN, BK, DOUT, 5u, 3u);
    };

    if (c.wave == 0) {
        issue(0, 0);
        wait_tensor0();
    }
    __syncthreads();

    const int NKT = DIN / BK;            // 128 k-steps
    for (int kt = 0; kt < NKT; ++kt) {
        const int s = kt & 1;
        if (c.wave == 0 && kt + 1 < NKT)
            issue(kt + 1, s ^ 1);        // async fill of the other stage
        gemm_mma_step(c, sxh[s], sxl[s], swh[s], swl[s], acc);
        if (c.wave == 0 && kt + 1 < NKT)
            wait_tensor0();              // next stage resident before release
        __syncthreads();
    }
    gemm_epilogue(c, acc, bias, loraB, txA, out);
}

// ---------------------------------------------------------------------------
// Fallback GEMM: in-kernel fp32->bf16 convert + LDS transpose (round-1 path).
// ---------------------------------------------------------------------------
__device__ __forceinline__ void stage_tiles(unsigned short* sxh, unsigned short* sxl,
                                            unsigned short* swh, unsigned short* swl,
                                            const float4* xv, const float4* wv,
                                            int ldrow, int ldk)
{
    #pragma unroll
    for (int i = 0; i < 4; ++i) {
        const int row = ldrow + 32 * i;
        float fx[4] = {xv[i].x, xv[i].y, xv[i].z, xv[i].w};
        unsigned short h[4], l[4];
        #pragma unroll
        for (int j = 0; j < 4; ++j) {
            h[j] = f2bf(fx[j]);
            l[j] = f2bf(fx[j] - bf2f(h[j]));
        }
        unsigned int* ph = (unsigned int*)(sxh + row * LDXK + ldk);
        unsigned int* pl = (unsigned int*)(sxl + row * LDXK + ldk);
        ph[0] = (unsigned int)h[0] | ((unsigned int)h[1] << 16);
        ph[1] = (unsigned int)h[2] | ((unsigned int)h[3] << 16);
        pl[0] = (unsigned int)l[0] | ((unsigned int)l[1] << 16);
        pl[1] = (unsigned int)l[2] | ((unsigned int)l[3] << 16);

        float fw[4] = {wv[i].x, wv[i].y, wv[i].z, wv[i].w};
        #pragma unroll
        for (int j = 0; j < 4; ++j) {
            unsigned short wh = f2bf(fw[j]);
            unsigned short wl = f2bf(fw[j] - bf2f(wh));
            swh[(ldk + j) * LDWN + row] = wh;
            swl[(ldk + j) * LDWN + row] = wl;
        }
    }
}

__global__ __launch_bounds__(256)
void lora_gemm_kernel(const float* __restrict__ x,
                      const float* __restrict__ wgt,
                      const float* __restrict__ bias,
                      const float* __restrict__ loraB,
                      const float* __restrict__ txA,
                      float* __restrict__ out)
{
    __shared__ __align__(16) unsigned short sxh[BM * LDXK];
    __shared__ __align__(16) unsigned short sxl[BM * LDXK];
    __shared__ __align__(16) unsigned short swh[BK * LDWN];
    __shared__ __align__(16) unsigned short swl[BK * LDWN];

    WaveCtx c;
    c.tid = threadIdx.x; c.lane = c.tid & 31; c.wave = c.tid >> 5;
    c.wm = c.wave >> 2; c.wn = c.wave & 3;
    c.bm0 = blockIdx.y * BM; c.bn0 = blockIdx.x * BN;

    v8f acc[4][2];
    #pragma unroll
    for (int mi = 0; mi < 4; ++mi)
        #pragma unroll
        for (int ni = 0; ni < 2; ++ni)
            #pragma unroll
            for (int e = 0; e < 8; ++e)
                acc[mi][ni][e] = 0.0f;

    const int ldrow = c.tid >> 3;
    const int ldk   = (c.tid & 7) << 2;
    const float* xg = x   + (size_t)(c.bm0 + ldrow) * DIN + ldk;
    const float* wg = wgt + (size_t)(c.bn0 + ldrow) * DIN + ldk;

    float4 xv[4], wv[4];
    #pragma unroll
    for (int i = 0; i < 4; ++i) {
        xv[i] = *(const float4*)(xg + (size_t)(32 * i) * DIN);
        wv[i] = *(const float4*)(wg + (size_t)(32 * i) * DIN);
    }
    stage_tiles(sxh, sxl, swh, swl, xv, wv, ldrow, ldk);
    __syncthreads();

    const int NKT = DIN / BK;
    for (int kt = 0; kt < NKT; ++kt) {
        if (kt + 1 < NKT) {
            const int koff = (kt + 1) * BK;
            #pragma unroll
            for (int i = 0; i < 4; ++i) {
                xv[i] = *(const float4*)(xg + (size_t)(32 * i) * DIN + koff);
                wv[i] = *(const float4*)(wg + (size_t)(32 * i) * DIN + koff);
            }
        }
        gemm_mma_step(c, sxh, sxl, swh, swl, acc);
        __syncthreads();
        if (kt + 1 < NKT)
            stage_tiles(sxh, sxl, swh, swl, xv, wv, ldrow, ldk);
        __syncthreads();
    }
    gemm_epilogue(c, acc, bias, loraB, txA, out);
}

// ---------------------------------------------------------------------------
extern "C" void kernel_launch(void* const* d_in, const int* in_sizes, int n_in,
                              void* d_out, int out_size, void* d_ws, size_t ws_size,
                              hipStream_t stream)
{
    const float* x     = (const float*)d_in[0];
    const float* wgt   = (const float*)d_in[1];
    const float* bias  = (const float*)d_in[2];
    const float* loraA = (const float*)d_in[3];
    const float* loraB = (const float*)d_in[4];
    float* out = (float*)d_out;

    const size_t TXA_BYTES = (size_t)TOKENS * RANK * sizeof(float);   // 512 KB
    const size_t XB = (size_t)TOKENS * DIN * sizeof(unsigned short);  // 64 MB
    const size_t WB = (size_t)DIN * DOUT * sizeof(unsigned short);    // 32 MB
    const size_t need = TXA_BYTES + 2 * XB + 2 * WB;                  // ~192.5 MB

    char* ws = (char*)d_ws;
    float* txA = (float*)ws;

    dim3 grid(DOUT / BN, TOKENS / BM);   // (32, 64)

    lora_xa_kernel<<<TOKENS / 8, 256, 0, stream>>>(x, loraA, txA);

    if (ws_size >= need) {
        unsigned short* xh  = (unsigned short*)(ws + TXA_BYTES);
        unsigned short* xl  = (unsigned short*)(ws + TXA_BYTES + XB);
        unsigned short* wth = (unsigned short*)(ws + TXA_BYTES + 2 * XB);
        unsigned short* wtl = (unsigned short*)(ws + TXA_BYTES + 2 * XB + WB);

        convert_x_kernel<<<(TOKENS * (DIN / 4)) / 256, 256, 0, stream>>>(x, xh, xl);
        convert_wT_kernel<<<dim3(DOUT / TT, DIN / TT), 256, 0, stream>>>(wgt, wth, wtl);
        lora_gemm_tdm_kernel<<<grid, 256, 0, stream>>>(xh, xl, wth, wtl,
                                                       bias, loraB, txA, out);
    } else {
        lora_gemm_kernel<<<grid, 256, 0, stream>>>(x, wgt, bias, loraB, txA, out);
    }
}
